// MixtureAttention_13735305412672
// MI455X (gfx1250) — matrix operand order repaired and checked by reading, not compiled, not run on hardware
//
#include <hip/hip_runtime.h>
#include <hip/hip_bf16.h>
#include <math.h>

#define HIDDEN 512
#define EN 256
#define ET 512
#define XDIM 768          // EN+ET
#define KCOMB 1280        // XDIM + HIDDEN
#define G4 2048           // 4*HIDDEN
#define BATCH 32
#define SEQL 128
#define ATTNW 50
#define VOCAB_N 300
#define VOCAB_T 10000
#define V_OUT 10053       // VOCAB_T + ATTN + 3
#define NPAD 10080        // V_OUT rounded up to mult of 32
#define EOF_N 299
#define EOF_T 9999
#define CONF 0.9f

typedef __bf16 bf16;
typedef __attribute__((ext_vector_type(16))) __bf16 v16bf;
typedef __attribute__((ext_vector_type(8)))  __bf16 v8bf;
typedef __attribute__((ext_vector_type(8)))  float  v8f;

// ---- workspace layout (bytes, all bf16 regions 32B-aligned) ----
static constexpr size_t OFF_WC = 0;                                       // 2048*1280 bf16
static constexpr size_t OFF_WH = OFF_WC + (size_t)G4 * KCOMB * 2;         // 512*512 bf16
static constexpr size_t OFF_WG = OFF_WH + (size_t)HIDDEN * HIDDEN * 2;    // 10080*1536 bf16
static constexpr size_t OFF_HS = OFF_WG + (size_t)NPAD * (3 * HIDDEN) * 2;// B*L*512 f32
static constexpr size_t OFF_H  = OFF_HS + (size_t)BATCH * SEQL * HIDDEN * 4;
static constexpr size_t OFF_C  = OFF_H + (size_t)BATCH * HIDDEN * 4;
static constexpr size_t OFF_XC = OFF_C + (size_t)BATCH * HIDDEN * 4;      // B*1280 bf16
static constexpr size_t OFF_GT = OFF_XC + (size_t)BATCH * KCOMB * 2;      // B*2048 f32
static constexpr size_t OFF_Q  = OFF_GT + (size_t)BATCH * G4 * 4;         // B*512 f32
static constexpr size_t OFF_AB = OFF_Q + (size_t)BATCH * HIDDEN * 4;      // B*1536 bf16
static constexpr size_t OFF_AT = OFF_AB + (size_t)BATCH * 3 * HIDDEN * 2; // B*50 f32 (pad 64)
static constexpr size_t OFF_LG = OFF_AT + (size_t)BATCH * 64 * 4;         // B*10080 f32

// ================= init / conversion =================
__global__ void init_kernel(float* zbuf, size_t n, float* out) {
    size_t i = (size_t)blockIdx.x * blockDim.x + threadIdx.x;
    size_t stride = (size_t)gridDim.x * blockDim.x;
    for (; i < n; i += stride) zbuf[i] = 0.0f;
    if (blockIdx.x == 0 && threadIdx.x == 0) out[0] = 0.0f;
}

__global__ void conv_wc(const float* __restrict__ w_ih, const float* __restrict__ w_hh,
                        bf16* __restrict__ wc) {
    size_t n = (size_t)G4 * KCOMB;
    size_t i = (size_t)blockIdx.x * blockDim.x + threadIdx.x;
    size_t stride = (size_t)gridDim.x * blockDim.x;
    for (; i < n; i += stride) {
        int g = (int)(i / KCOMB), k = (int)(i % KCOMB);
        float v = (k < XDIM) ? w_ih[(size_t)g * XDIM + k] : w_hh[(size_t)g * HIDDEN + (k - XDIM)];
        wc[i] = (bf16)v;
    }
}

__global__ void conv_wh(const float* __restrict__ wh, bf16* __restrict__ o) {
    size_t n = (size_t)HIDDEN * HIDDEN;
    size_t i = (size_t)blockIdx.x * blockDim.x + threadIdx.x;
    size_t stride = (size_t)gridDim.x * blockDim.x;
    for (; i < n; i += stride) o[i] = (bf16)wh[i];
}

__global__ void conv_wg(const float* __restrict__ wg, bf16* __restrict__ o) {
    size_t n = (size_t)NPAD * (3 * HIDDEN);
    size_t i = (size_t)blockIdx.x * blockDim.x + threadIdx.x;
    size_t stride = (size_t)gridDim.x * blockDim.x;
    for (; i < n; i += stride) {
        int r = (int)(i / (3 * HIDDEN));
        o[i] = (r < V_OUT) ? (bf16)wg[i] : (bf16)0.0f;
    }
}

// ================= per-step kernels =================
__global__ void build_xc(const int* __restrict__ n_t, const int* __restrict__ t_t,
                         const float* __restrict__ embN, const float* __restrict__ embT,
                         const float* __restrict__ h, bf16* __restrict__ xc, int t) {
    int i = blockIdx.x * blockDim.x + threadIdx.x;
    if (i >= BATCH * KCOMB) return;
    int b = i / KCOMB, k = i % KCOMB;
    float v;
    if (k < EN) {
        int in_n = (t > 0) ? n_t[b * SEQL + (t - 1)] : 0;
        v = embN[(size_t)in_n * EN + k];
    } else if (k < XDIM) {
        int in_t = (t > 0) ? t_t[b * SEQL + (t - 1)] : 0;
        v = embT[(size_t)in_t * ET + (k - EN)];
    } else {
        v = h[(size_t)b * HIDDEN + (k - XDIM)];
    }
    xc[i] = (bf16)v;
}

#define WMMA_BF16(a, b, c) \
    __builtin_amdgcn_wmma_f32_16x16x32_bf16(false, (a), false, (b), (short)0, (c), false, false)

// D(32xN) = A(32xK) * B(NxK)^T + bias
// One wave per 32x32 output block: 2 M-tiles x 2 N-tiles -> 4 WMMAs per K-chunk,
// A and B registers each reused twice (2 b128 loads per WMMA).
__global__ __launch_bounds__(32)
void wmma_gemm_bf16(const bf16* __restrict__ A, const bf16* __restrict__ Bm,
                    float* __restrict__ Cout,
                    const float* __restrict__ bias1, const float* __restrict__ bias2,
                    int K, int lda, int ldb, int ldc, int biasN) {
    int ngrp = blockIdx.x;   // 32-wide N group
    int lane = threadIdx.x;
    int half = lane >> 4;
    int l15  = lane & 15;
    const bf16* arow0 = A + (size_t)l15 * lda;          // M rows 0..15
    const bf16* arow1 = A + (size_t)(16 + l15) * lda;   // M rows 16..31
    const bf16* brow0 = Bm + (size_t)(ngrp * 32 + l15) * ldb;        // N cols 0..15
    const bf16* brow1 = Bm + (size_t)(ngrp * 32 + 16 + l15) * ldb;   // N cols 16..31
    v8f acc00 = {}, acc01 = {}, acc10 = {}, acc11 = {};
    for (int kc = 0; kc < K; kc += 32) {
        // A 16x32 bf16 layout: half 0 -> K kc..kc+15; half 1 -> K kc+16..kc+31
        v8bf a0lo = *(const v8bf*)(arow0 + kc + 16 * half);
        v8bf a0hi = *(const v8bf*)(arow0 + kc + 16 * half + 8);
        v8bf a1lo = *(const v8bf*)(arow1 + kc + 16 * half);
        v8bf a1hi = *(const v8bf*)(arow1 + kc + 16 * half + 8);
        v16bf a0, a1;
#pragma unroll
        for (int i = 0; i < 8; ++i) {
            a0[i] = a0lo[i]; a0[i + 8] = a0hi[i];
            a1[i] = a1lo[i]; a1[i + 8] = a1hi[i];
        }
        // B 32x16 bf16 layout: lanes 0-15 K 0..15, lanes 16-31 K 16..31 (col = lane&15)
        v16bf b0 = *(const v16bf*)(brow0 + kc + 16 * half);
        v16bf b1 = *(const v16bf*)(brow1 + kc + 16 * half);
        acc00 = WMMA_BF16(a0, b0, acc00);
        acc01 = WMMA_BF16(a0, b1, acc01);
        acc10 = WMMA_BF16(a1, b0, acc10);
        acc11 = WMMA_BF16(a1, b1, acc11);
    }
    int n0 = ngrp * 32 + l15;
    int n1 = n0 + 16;
    float bn0 = 0.0f, bn1 = 0.0f;
    if (bias1) {
        if (n0 < biasN) bn0 += bias1[n0];
        if (n1 < biasN) bn1 += bias1[n1];
    }
    if (bias2) {
        if (n0 < biasN) bn0 += bias2[n0];
        if (n1 < biasN) bn1 += bias2[n1];
    }
#pragma unroll
    for (int r = 0; r < 8; ++r) {
        int m0 = r + 8 * half;
        int m1 = 16 + r + 8 * half;
        Cout[(size_t)m0 * ldc + n0] = acc00[r] + bn0;
        Cout[(size_t)m0 * ldc + n1] = acc01[r] + bn1;
        Cout[(size_t)m1 * ldc + n0] = acc10[r] + bn0;
        Cout[(size_t)m1 * ldc + n1] = acc11[r] + bn1;
    }
}

__device__ __forceinline__ float sigf(float x) { return 1.0f / (1.0f + expf(-x)); }

__global__ void lstm_update(const float* __restrict__ gates, float* __restrict__ h,
                            float* __restrict__ c, float* __restrict__ hs,
                            const int* __restrict__ p_t, bf16* __restrict__ abuf, int t) {
    int i = blockIdx.x * blockDim.x + threadIdx.x;
    if (i >= BATCH * HIDDEN) return;
    int b = i / HIDDEN, j = i % HIDDEN;
    const float* g = gates + (size_t)b * G4;
    float ig = sigf(g[j]);
    float fg = sigf(g[HIDDEN + j]);
    float gg = tanhf(g[2 * HIDDEN + j]);
    float og = sigf(g[3 * HIDDEN + j]);
    float cn = fg * c[i] + ig * gg;
    float hn = og * tanhf(cn);
    int parent = (t > 0) ? p_t[b * SEQL + (t - 1)] : 0;
    float hp = hs[((size_t)b * SEQL + parent) * HIDDEN + j];  // read old hs before write
    hs[((size_t)b * SEQL + t) * HIDDEN + j] = hn;
    h[i] = hn; c[i] = cn;
    bf16* ab = abuf + (size_t)b * (3 * HIDDEN);
    ab[j] = (bf16)hn;
    ab[HIDDEN + j] = (bf16)cn;
    ab[2 * HIDDEN + j] = (bf16)hp;
}

__global__ __launch_bounds__(256)
void attn_kernel(const float* __restrict__ q, const float* __restrict__ hs,
                 const int* __restrict__ n_t, const float* __restrict__ v_w,
                 const float* __restrict__ v_b, float* __restrict__ attn, int t) {
    int b = blockIdx.x;
    int lane = threadIdx.x & 31;
    int wave = threadIdx.x >> 5;
    __shared__ float sc[ATTNW];
    for (int w = wave; w < ATTNW; w += 8) {
        int idx = t - ATTNW + w;
        bool valid = idx >= 0;
        int idxc = valid ? idx : 0;
        bool masked = (!valid) || (n_t[b * SEQL + idxc] == EOF_N);
        float s;
        if (masked) {
            s = -1e20f;
        } else {
            const float* mem = hs + ((size_t)b * SEQL + idxc) * HIDDEN;
            const float* qb  = q + (size_t)b * HIDDEN;
            float p = 0.0f;
            for (int k = lane; k < HIDDEN; k += 32) p += tanhf(qb[k] + mem[k]) * v_w[k];
            for (int off = 16; off > 0; off >>= 1) p += __shfl_down(p, off, 32);
            s = p + v_b[0];
        }
        if (lane == 0) sc[w] = s;
    }
    __syncthreads();
    if (threadIdx.x == 0) {
        float m = sc[0];
        for (int w = 1; w < ATTNW; ++w) m = fmaxf(m, sc[w]);
        float Z = 0.0f;
        for (int w = 0; w < ATTNW; ++w) Z += expf(sc[w] - m);
        for (int w = 0; w < ATTNW; ++w) attn[b * ATTNW + w] = expf(sc[w] - m) / Z;
    }
}

__global__ __launch_bounds__(256)
void finalize_kernel(const float* __restrict__ logits, const float* __restrict__ attn,
                     const float* __restrict__ h, const float* __restrict__ c,
                     const float* __restrict__ Ws_w, const float* __restrict__ Ws_b,
                     const int* __restrict__ t_t, float* __restrict__ out, int t) {
    int b = blockIdx.x, tid = threadIdx.x;
    const float* lg = logits + (size_t)b * NPAD;
    __shared__ float red[256];
    __shared__ int redi[256];
    // s_t = sigmoid([h,c] . Ws_w + Ws_b)
    float p = 0.0f;
    for (int k = tid; k < 2 * HIDDEN; k += 256) {
        float v = (k < HIDDEN) ? h[(size_t)b * HIDDEN + k]
                               : c[(size_t)b * HIDDEN + (k - HIDDEN)];
        p += v * Ws_w[k];
    }
    red[tid] = p; __syncthreads();
    for (int s = 128; s > 0; s >>= 1) {
        if (tid < s) red[tid] += red[tid + s];
        __syncthreads();
    }
    float s_t = sigf(red[0] + Ws_b[0]); __syncthreads();
    // row max
    float m = -INFINITY;
    for (int j = tid; j < V_OUT; j += 256) m = fmaxf(m, lg[j]);
    red[tid] = m; __syncthreads();
    for (int s = 128; s > 0; s >>= 1) {
        if (tid < s) red[tid] = fmaxf(red[tid], red[tid + s]);
        __syncthreads();
    }
    m = red[0]; __syncthreads();
    // sum exp
    float Z = 0.0f;
    for (int j = tid; j < V_OUT; j += 256) Z += expf(lg[j] - m);
    red[tid] = Z; __syncthreads();
    for (int s = 128; s > 0; s >>= 1) {
        if (tid < s) red[tid] += red[tid + s];
        __syncthreads();
    }
    float logZ = m + logf(red[0]); __syncthreads();
    // argmax of s_t*(log_softmax)
    float bm = -INFINITY; int bi = 0;
    for (int j = tid; j < V_OUT; j += 256) {
        float v = s_t * (lg[j] - logZ);
        if (v > bm) { bm = v; bi = j; }
    }
    red[tid] = bm; redi[tid] = bi; __syncthreads();
    for (int s = 128; s > 0; s >>= 1) {
        if (tid < s) {
            if (red[tid + s] > red[tid] ||
                (red[tid + s] == red[tid] && redi[tid + s] < redi[tid])) {
                red[tid] = red[tid + s]; redi[tid] = redi[tid + s];
            }
        }
        __syncthreads();
    }
    float maxOut = red[0]; int idxOut = redi[0]; __syncthreads();

    __shared__ int scond;
    if (tid == 0) {
        float ml = -INFINITY; int mi = 0;
        for (int w = 0; w < ATTNW; ++w) {
            float v = (1.0f - s_t) * attn[b * ATTNW + w];
            if (v > ml) { ml = v; mi = w; }
        }
        scond = (maxOut > ml) ? 1 : 0;
        int topi = scond ? (VOCAB_T + mi) : idxOut;
        out[1 + b * SEQL + t] = (float)topi;
    }
    __syncthreads();
    int cond = scond;
    int tcol = t_t[b * SEQL + t];
    if (tcol != EOF_T) {
        const float smv = 0.1f / (float)(V_OUT - 2);
        const float lsmv = logf(smv);
        const float lconf = logf(CONF);
        float acc = 0.0f;
        for (int j = tid; j < V_OUT; j += 256) {
            if (j == EOF_T) continue;
            float mp, lmp;
            if (j == tcol) { mp = CONF; lmp = lconf; }
            else { mp = smv; lmp = lsmv; }
            float ov;
            if (cond) {
                int w = j - VOCAB_T;
                ov = (w >= 0 && w < ATTNW) ? (1.0f - s_t) * attn[b * ATTNW + w] : 0.0f;
            } else {
                ov = s_t * (lg[j] - logZ);
            }
            acc += mp * (lmp - ov);
        }
        red[tid] = acc; __syncthreads();
        for (int s = 128; s > 0; s >>= 1) {
            if (tid < s) red[tid] += red[tid + s];
            __syncthreads();
        }
        if (tid == 0) atomicAdd(out, red[0]);
    }
}

// ================= launcher =================
extern "C" void kernel_launch(void* const* d_in, const int* in_sizes, int n_in,
                              void* d_out, int out_size, void* d_ws, size_t ws_size,
                              hipStream_t stream) {
    (void)in_sizes; (void)n_in; (void)out_size; (void)ws_size;
    const int*   n_t  = (const int*)d_in[0];
    const int*   t_t  = (const int*)d_in[1];
    const int*   p_t  = (const int*)d_in[2];
    const float* embN = (const float*)d_in[3];
    const float* embT = (const float*)d_in[4];
    const float* w_ih = (const float*)d_in[5];
    const float* w_hh = (const float*)d_in[6];
    const float* b_ih = (const float*)d_in[7];
    const float* b_hh = (const float*)d_in[8];
    const float* Wh_w = (const float*)d_in[9];
    const float* Wh_b = (const float*)d_in[10];
    const float* v_w  = (const float*)d_in[11];
    const float* v_b  = (const float*)d_in[12];
    const float* Wg_w = (const float*)d_in[13];
    const float* Wg_b = (const float*)d_in[14];
    const float* Ws_w = (const float*)d_in[15];
    const float* Ws_b = (const float*)d_in[16];
    float* out = (float*)d_out;
    char*  ws  = (char*)d_ws;

    bf16*  WC = (bf16*)(ws + OFF_WC);
    bf16*  WH = (bf16*)(ws + OFF_WH);
    bf16*  WG = (bf16*)(ws + OFF_WG);
    float* HS = (float*)(ws + OFF_HS);
    float* H  = (float*)(ws + OFF_H);
    float* C  = (float*)(ws + OFF_C);
    bf16*  XC = (bf16*)(ws + OFF_XC);
    float* GT = (float*)(ws + OFF_GT);
    float* Q  = (float*)(ws + OFF_Q);
    bf16*  AB = (bf16*)(ws + OFF_AB);
    float* AT = (float*)(ws + OFF_AT);
    float* LG = (float*)(ws + OFF_LG);

    // one-time per launch: zero state (hs, h, c are contiguous) + convert weights to bf16
    size_t nzero = (size_t)BATCH * SEQL * HIDDEN + 2 * (size_t)BATCH * HIDDEN;
    init_kernel<<<2048, 256, 0, stream>>>(HS, nzero, out);
    conv_wc<<<2048, 256, 0, stream>>>(w_ih, w_hh, WC);
    conv_wh<<<512, 256, 0, stream>>>(Wh_w, WH);
    conv_wg<<<4096, 256, 0, stream>>>(Wg_w, WG);

    for (int t = 0; t < SEQL; ++t) {
        build_xc<<<(BATCH * KCOMB + 255) / 256, 256, 0, stream>>>(n_t, t_t, embN, embT, H, XC, t);
        // gates = [x,h] @ [w_ih|w_hh]^T + b_ih + b_hh   (M=32,K=1280,N=2048)
        wmma_gemm_bf16<<<G4 / 32, 32, 0, stream>>>(XC, WC, GT, b_ih, b_hh,
                                                   KCOMB, KCOMB, KCOMB, G4, G4);
        lstm_update<<<(BATCH * HIDDEN + 255) / 256, 256, 0, stream>>>(GT, H, C, HS, p_t, AB, t);
        // q = h_new @ Wh^T + Wh_b   (M=32,K=512,N=512); A = h part of abuf (lda=1536)
        wmma_gemm_bf16<<<HIDDEN / 32, 32, 0, stream>>>(AB, WH, Q, Wh_b, nullptr,
                                                       HIDDEN, 3 * HIDDEN, HIDDEN,
                                                       HIDDEN, HIDDEN);
        attn_kernel<<<BATCH, 256, 0, stream>>>(Q, HS, n_t, v_w, v_b, AT, t);
        // logits = [h,c,hp] @ Wg^T + Wg_b   (M=32,K=1536,N=10080 padded)
        wmma_gemm_bf16<<<NPAD / 32, 32, 0, stream>>>(AB, WG, LG, Wg_b, nullptr,
                                                     3 * HIDDEN, 3 * HIDDEN, 3 * HIDDEN,
                                                     NPAD, V_OUT);
        finalize_kernel<<<BATCH, 256, 0, stream>>>(LG, AT, H, C, Ws_w, Ws_b, t_t, out, t);
    }
}
